// BasicTransformerBlock_89962384982092
// MI455X (gfx1250) — matrix operand order repaired and checked by reading, not compiled, not run on hardware
//
#include <hip/hip_runtime.h>

// ============================================================================
// BasicTransformerBlock for MI455X (gfx1250, wave32, WMMA).
// bf16 A/B operands, fp32 accumulation (v_wmma_f32_16x16x32_bf16).
// This revision software-pipelines all WMMA loops (double-buffered fragments)
// so global loads overlap matrix ops instead of s_wait_loadcnt 0x0 stalls,
// and fixes the prefetch scope (was SCOPE_SYS -> near-cache).
// ============================================================================

typedef __attribute__((ext_vector_type(16))) __bf16 v16bf;
typedef __attribute__((ext_vector_type(8)))  __bf16 v8bf;
typedef __attribute__((ext_vector_type(4)))  __bf16 v4bf;
typedef __attribute__((ext_vector_type(8)))  float  v8f;

#define DEV __device__ __forceinline__

DEV v8f vzero8() {
  v8f z;
#pragma unroll
  for (int i = 0; i < 8; ++i) z[i] = 0.0f;
  return z;
}

DEV v8f wmma_bf16(v16bf a, v16bf b, v8f c) {
  return __builtin_amdgcn_wmma_f32_16x16x32_bf16(false, a, false, b, (short)0, c,
                                                 false, false);
}

// A fragment 16x32 (MxK) from row-major [*, lda] bf16.
// lane<16: M=lane, K=0..15; lane>=16: M=lane-16, K=16..31
DEV v16bf load_afrag(const __bf16* A, int lda, int row, int k0, int lane) {
  int off = (lane >> 4) ? 16 : 0;
  const __bf16* p = A + (size_t)row * lda + k0 + off;
  v8bf lo = *(const v8bf*)(p);
  v8bf hi = *(const v8bf*)(p + 8);
  return __builtin_shufflevector(lo, hi, 0, 1, 2, 3, 4, 5, 6, 7, 8, 9, 10, 11,
                                 12, 13, 14, 15);
}

// B fragment 32x16 (KxN) from a [N][K] row-major array.
// lane<16: N=lane, K=0..15 ; lane>=16: N=lane-16, K=16..31  (contiguous 32B)
DEV v16bf load_bfrag(const __bf16* Bt, int ldb, int col, int k0, int lane) {
  int off = (lane >> 4) ? 16 : 0;
  return *(const v16bf*)(Bt + (size_t)col * ldb + k0 + off);
}

// ---------------------------------------------------------------------------
// Pipelined GEMM core: acc[t] += sum_k A[arow,k] * W[n0+t*16+lrow, k]
// Double-buffers A + 4 B fragments: loads for block k+32 issue before the
// WMMAs consuming block k, so VMEM latency hides behind the matrix pipe.
// ---------------------------------------------------------------------------
DEV void gemm_core(const __bf16* __restrict__ A, const __bf16* __restrict__ W,
                   int K, int arow, int n0, int lane, v8f acc[4]) {
  int lrow = lane & 15;
  v16bf a_cur = load_afrag(A, K, arow, 0, lane);
  v16bf b_cur[4];
#pragma unroll
  for (int t = 0; t < 4; ++t)
    b_cur[t] = load_bfrag(W, K, n0 + t * 16 + lrow, 0, lane);

  for (int k0 = 32; k0 < K; k0 += 32) {
    __builtin_prefetch(A + (size_t)arow * K + k0 + 32, 0, 3);
    v16bf a_nxt = load_afrag(A, K, arow, k0, lane);
    v16bf b_nxt[4];
#pragma unroll
    for (int t = 0; t < 4; ++t)
      b_nxt[t] = load_bfrag(W, K, n0 + t * 16 + lrow, k0, lane);
#pragma unroll
    for (int t = 0; t < 4; ++t) acc[t] = wmma_bf16(a_cur, b_cur[t], acc[t]);
    a_cur = a_nxt;
#pragma unroll
    for (int t = 0; t < 4; ++t) b_cur[t] = b_nxt[t];
  }
#pragma unroll
  for (int t = 0; t < 4; ++t) acc[t] = wmma_bf16(a_cur, b_cur[t], acc[t]);
}

// ---------------------------------------------------------------------------
// fp32 -> bf16 elementwise (weights)
// ---------------------------------------------------------------------------
__global__ void f32_to_bf16(const float* __restrict__ s, __bf16* __restrict__ d,
                            int n) {
  int i = blockIdx.x * 256 + threadIdx.x;
  if (i < n) d[i] = (__bf16)s[i];
}

// ---------------------------------------------------------------------------
// [b,C,N] fp32 -> token-major [b*N, C] (fp32 or bf16), tiled LDS transpose
// ---------------------------------------------------------------------------
__global__ void transpose_bcn_f32(const float* __restrict__ X,
                                  float* __restrict__ Y, int C, int N) {
  __shared__ float tile[32][33];
  int b = blockIdx.z;
  int n0 = blockIdx.x * 32, c0 = blockIdx.y * 32;
  const float* src = X + (size_t)b * C * N;
#pragma unroll
  for (int i = threadIdx.y; i < 32; i += 8)
    tile[i][threadIdx.x] = src[(size_t)(c0 + i) * N + n0 + threadIdx.x];
  __syncthreads();
  float* dst = Y + (size_t)b * N * C;
#pragma unroll
  for (int i = threadIdx.y; i < 32; i += 8)
    dst[(size_t)(n0 + i) * C + c0 + threadIdx.x] = tile[threadIdx.x][i];
}

__global__ void transpose_bcn_bf16(const float* __restrict__ X,
                                   __bf16* __restrict__ Y, int C, int N) {
  __shared__ float tile[32][33];
  int b = blockIdx.z;
  int n0 = blockIdx.x * 32, c0 = blockIdx.y * 32;
  const float* src = X + (size_t)b * C * N;
#pragma unroll
  for (int i = threadIdx.y; i < 32; i += 8)
    tile[i][threadIdx.x] = src[(size_t)(c0 + i) * N + n0 + threadIdx.x];
  __syncthreads();
  __bf16* dst = Y + (size_t)b * N * C;
#pragma unroll
  for (int i = threadIdx.y; i < 32; i += 8)
    dst[(size_t)(n0 + i) * C + c0 + threadIdx.x] = (__bf16)tile[threadIdx.x][i];
}

// ---------------------------------------------------------------------------
// Channel LayerNorm over C=512, token-major input. One wave per token.
// ---------------------------------------------------------------------------
__global__ void ln512_tokmajor(const float* __restrict__ X,
                               const float* __restrict__ g,
                               __bf16* __restrict__ H) {
  int wave = threadIdx.x >> 5;
  int lane = threadIdx.x & 31;
  int t = blockIdx.x * 8 + wave;
  const float* row = X + (size_t)t * 512;
  float4 v[4];
  float s = 0.f, ss = 0.f;
#pragma unroll
  for (int j = 0; j < 4; ++j) {
    v[j] = *(const float4*)(row + lane * 4 + j * 128);
    s += v[j].x + v[j].y + v[j].z + v[j].w;
    ss += v[j].x * v[j].x + v[j].y * v[j].y + v[j].z * v[j].z + v[j].w * v[j].w;
  }
#pragma unroll
  for (int m = 1; m < 32; m <<= 1) {
    s += __shfl_xor(s, m, 32);
    ss += __shfl_xor(ss, m, 32);
  }
  float mean = s * (1.0f / 512.0f);
  float var = ss * (1.0f / 512.0f) - mean * mean;
  float rs = rsqrtf(var + 1e-5f);
  __bf16* orow = H + (size_t)t * 512;
#pragma unroll
  for (int j = 0; j < 4; ++j) {
    int c = lane * 4 + j * 128;
    float4 gj = *(const float4*)(g + c);
    v4bf o;
    o[0] = (__bf16)((v[j].x - mean) * rs * gj.x);
    o[1] = (__bf16)((v[j].y - mean) * rs * gj.y);
    o[2] = (__bf16)((v[j].z - mean) * rs * gj.z);
    o[3] = (__bf16)((v[j].w - mean) * rs * gj.w);
    *(v4bf*)(orow + c) = o;
  }
}

// ---------------------------------------------------------------------------
// GEMM kernels: wave = 16x64 C strip, block = 8 waves (64M x 128N macro-tile)
// ---------------------------------------------------------------------------
#define GEMM_PROLOGUE()                                              \
  int lane = threadIdx.x & 31;                                       \
  int wave = threadIdx.x >> 5;                                       \
  int lrow = lane & 15, hi = lane >> 4;                              \
  int m0 = blockIdx.x * 64 + (wave & 3) * 16;                        \
  int n0 = blockIdx.y * 128 + (wave >> 2) * 64;                      \
  int arow = m0 + lrow;                                              \
  v8f acc[4];                                                        \
  _Pragma("unroll") for (int t = 0; t < 4; ++t) acc[t] = vzero8();

// out bf16 token-major
__global__ void gemm_bf16_out(const __bf16* __restrict__ A,
                              const __bf16* __restrict__ W,
                              __bf16* __restrict__ C, int Ncols, int K) {
  GEMM_PROLOGUE();
  gemm_core(A, W, K, arow, n0, lane, acc);
#pragma unroll
  for (int t = 0; t < 4; ++t)
#pragma unroll
    for (int v = 0; v < 8; ++v)
      C[(size_t)(m0 + v + 8 * hi) * Ncols + n0 + t * 16 + lrow] =
          (__bf16)acc[t][v];
}

// V projection: write transposed  VT[(b*512 + c)*tokb + tok_in_batch]
__global__ void gemm_vt(const __bf16* __restrict__ A,
                        const __bf16* __restrict__ W, __bf16* __restrict__ VT,
                        int K, int tokb) {
  GEMM_PROLOGUE();
  gemm_core(A, W, K, arow, n0, lane, acc);
#pragma unroll
  for (int t = 0; t < 4; ++t)
#pragma unroll
    for (int v = 0; v < 8; ++v) {
      int r = m0 + v + 8 * hi;
      int c = n0 + t * 16 + lrow;
      int b = r / tokb, tin = r - b * tokb;
      VT[((size_t)b * 512 + c) * tokb + tin] = (__bf16)acc[t][v];
    }
}

// Output projection: X[r,c] += acc + bias[c]   (fp32 running activation)
__global__ void gemm_f32_bias_res(const __bf16* __restrict__ A,
                                  const __bf16* __restrict__ W,
                                  const float* __restrict__ bias,
                                  float* __restrict__ X, int Ncols, int K) {
  GEMM_PROLOGUE();
  gemm_core(A, W, K, arow, n0, lane, acc);
#pragma unroll
  for (int t = 0; t < 4; ++t)
#pragma unroll
    for (int v = 0; v < 8; ++v) {
      int r = m0 + v + 8 * hi;
      int c = n0 + t * 16 + lrow;
      X[(size_t)r * Ncols + c] += acc[t][v] + bias[c];
    }
}

// Fused GEGLU first GEMM: two passes of the pipelined core (h and gate),
// epilogue h * gelu(gate) -> bf16 act.
__global__ void gemm_geglu(const __bf16* __restrict__ A,
                           const __bf16* __restrict__ W1,
                           const float* __restrict__ b1,
                           __bf16* __restrict__ ACT, int K) {
  GEMM_PROLOGUE();
  v8f ag[4];
#pragma unroll
  for (int t = 0; t < 4; ++t) ag[t] = vzero8();
  gemm_core(A, W1, K, arow, n0, lane, acc);          // h-half (cols n0..)
  gemm_core(A, W1, K, arow, n0 + 2048, lane, ag);    // gate-half
#pragma unroll
  for (int t = 0; t < 4; ++t)
#pragma unroll
    for (int v = 0; v < 8; ++v) {
      int r = m0 + v + 8 * hi;
      int c = n0 + t * 16 + lrow;
      float hv = acc[t][v] + b1[c];
      float gv = ag[t][v] + b1[c + 2048];
      float gelu = 0.5f * gv * (1.0f + erff(gv * 0.70710678118654752f));
      ACT[(size_t)r * 2048 + c] = (__bf16)(hv * gelu);
    }
}

// Final GEMM: OUT[b,c,n] = acc + b2[c] + X[r,c]   (fused residual + transpose)
__global__ void gemm_final(const __bf16* __restrict__ A,
                           const __bf16* __restrict__ W2,
                           const float* __restrict__ b2,
                           const float* __restrict__ X, float* __restrict__ OUT,
                           int K, int Ntok) {
  GEMM_PROLOGUE();
  gemm_core(A, W2, K, arow, n0, lane, acc);
#pragma unroll
  for (int t = 0; t < 4; ++t)
#pragma unroll
    for (int v = 0; v < 8; ++v) {
      int r = m0 + v + 8 * hi;
      int c = n0 + t * 16 + lrow;
      float val = acc[t][v] + b2[c] + X[(size_t)r * 512 + c];
      int b = r / Ntok, np = r - b * Ntok;
      OUT[((size_t)b * 512 + c) * Ntok + np] = val;
    }
}

// ---------------------------------------------------------------------------
// Flash attention.  Q,K token-major bf16 [B*tok, 512]; VT bf16 [(b*512+c)*NKb+j].
// One wave per 16 query rows, online softmax over 32-key blocks.
// Pipelining: V fragments issue right after the S WMMAs (their latency hides
// behind the softmax VALU + LDS bounce); next block's K fragments are
// preloaded with a wrap index (branch-free) before the P*V WMMAs.
// Grid (NQ/64, heads, B), block 128 (4 waves).
// ---------------------------------------------------------------------------
__global__ void flash_attn(const __bf16* __restrict__ Q,
                           const __bf16* __restrict__ K,
                           const __bf16* __restrict__ VT,
                           __bf16* __restrict__ O, int NQ, int NKb) {
  __shared__ __align__(16) __bf16 ldsP[4][16 * 32];
  int lane = threadIdx.x & 31;
  int wave = threadIdx.x >> 5;
  int lrow = lane & 15, hi = lane >> 4;
  int h = blockIdx.y, b = blockIdx.z;
  int i0 = blockIdx.x * 64 + wave * 16;
  int qrow = b * NQ + i0 + lrow;

  v16bf q0 = load_afrag(Q, 512, qrow, h * 64, lane);
  v16bf q1 = load_afrag(Q, 512, qrow, h * 64 + 32, lane);

  v8f acc[4];
#pragma unroll
  for (int t = 0; t < 4; ++t) acc[t] = vzero8();
  float rm[8], rl[8];
#pragma unroll
  for (int v = 0; v < 8; ++v) { rm[v] = -1e30f; rl[v] = 0.0f; }

  __bf16* myP = &ldsP[wave][0];
  const __bf16* vbase = VT + (size_t)b * 512 * NKb;
  const float SM = 0.125f * 1.44269504088896f;  // dh^-0.5 * log2(e)

  // preload K fragments for jb = 0
  int kcol = b * NKb + lrow;
  v16bf kb0 = load_bfrag(K, 512, kcol, h * 64, lane);
  v16bf kb1 = load_bfrag(K, 512, kcol, h * 64 + 32, lane);
  v16bf kc0 = load_bfrag(K, 512, kcol + 16, h * 64, lane);
  v16bf kc1 = load_bfrag(K, 512, kcol + 16, h * 64 + 32, lane);

  for (int jb = 0; jb < NKb; jb += 32) {
    v8f s0 = wmma_bf16(q0, kb0, vzero8());
    s0 = wmma_bf16(q1, kb1, s0);
    v8f s1 = wmma_bf16(q0, kc0, vzero8());
    s1 = wmma_bf16(q1, kc1, s1);

    // issue V fragment loads now; softmax VALU below hides their latency
    v16bf vf[4];
#pragma unroll
    for (int t = 0; t < 4; ++t) {
      int d = h * 64 + t * 16 + lrow;
      vf[t] = *(const v16bf*)(vbase + (size_t)d * NKb + jb + (hi ? 16 : 0));
    }
    // preload next block's K fragments (wrap index keeps this branch-free;
    // the extra loads on the last iteration are discarded)
    int jn = jb + 32;
    if (jn >= NKb) jn = 0;
    int kcoln = b * NKb + jn + lrow;
    kb0 = load_bfrag(K, 512, kcoln, h * 64, lane);
    kb1 = load_bfrag(K, 512, kcoln, h * 64 + 32, lane);
    kc0 = load_bfrag(K, 512, kcoln + 16, h * 64, lane);
    kc1 = load_bfrag(K, 512, kcoln + 16, h * 64 + 32, lane);

#pragma unroll
    for (int v = 0; v < 8; ++v) {
      float t0 = s0[v] * SM, t1 = s1[v] * SM;
      float tv = fmaxf(t0, t1);
#pragma unroll
      for (int m = 1; m < 16; m <<= 1) tv = fmaxf(tv, __shfl_xor(tv, m, 16));
      float newm = fmaxf(rm[v], tv);
      float corr = exp2f(rm[v] - newm);
      rm[v] = newm;
      float e0 = exp2f(t0 - newm);
      float e1 = exp2f(t1 - newm);
      float ts = e0 + e1;
#pragma unroll
      for (int m = 1; m < 16; m <<= 1) ts += __shfl_xor(ts, m, 16);
      rl[v] = rl[v] * corr + ts;
#pragma unroll
      for (int t = 0; t < 4; ++t) acc[t][v] *= corr;
      int r = v + 8 * hi;
      myP[r * 32 + lrow] = (__bf16)e0;       // cols jb..jb+15
      myP[r * 32 + 16 + lrow] = (__bf16)e1;  // cols jb+16..jb+31
    }

    // Re-read P in A-fragment layout (DS ops are in-order within a wave;
    // the dscnt scoreboard covers the VGPR dependency before the WMMA).
    int off = hi ? 16 : 0;
    const __bf16* pr = myP + lrow * 32 + off;
    v8bf plo = *(const v8bf*)(pr);
    v8bf phi = *(const v8bf*)(pr + 8);
    v16bf pf = __builtin_shufflevector(plo, phi, 0, 1, 2, 3, 4, 5, 6, 7, 8, 9,
                                       10, 11, 12, 13, 14, 15);
#pragma unroll
    for (int t = 0; t < 4; ++t) acc[t] = wmma_bf16(pf, vf[t], acc[t]);
  }

#pragma unroll
  for (int v = 0; v < 8; ++v) {
    float inv = 1.0f / rl[v];
    int tok = b * NQ + i0 + v + 8 * hi;
#pragma unroll
    for (int t = 0; t < 4; ++t)
      O[(size_t)tok * 512 + h * 64 + t * 16 + lrow] =
          (__bf16)(acc[t][v] * inv);
  }
}

// ===========================================================================
extern "C" void kernel_launch(void* const* d_in, const int* in_sizes, int n_in,
                              void* d_out, int out_size, void* d_ws,
                              size_t ws_size, hipStream_t stream) {
  const float* x = (const float*)d_in[0];
  const float* ctx = (const float*)d_in[1];
  const float* g1 = (const float*)d_in[2];
  const float* Wq1 = (const float*)d_in[3];
  const float* Wk1 = (const float*)d_in[4];
  const float* Wv1 = (const float*)d_in[5];
  const float* Wo1 = (const float*)d_in[6];
  const float* bo1 = (const float*)d_in[7];
  const float* g2 = (const float*)d_in[8];
  const float* Wq2 = (const float*)d_in[9];
  const float* Wk2 = (const float*)d_in[10];
  const float* Wv2 = (const float*)d_in[11];
  const float* Wo2 = (const float*)d_in[12];
  const float* bo2 = (const float*)d_in[13];
  const float* g3 = (const float*)d_in[14];
  const float* Wff1 = (const float*)d_in[15];
  const float* bff1 = (const float*)d_in[16];
  const float* Wff2 = (const float*)d_in[17];
  const float* bff2 = (const float*)d_in[18];
  float* OUT = (float*)d_out;

  const int Bb = 4, Nn = 2048, CTXC = 768, Mm = 256;
  const int T = Bb * Nn;   // 8192 x-tokens
  const int Tc = Bb * Mm;  // 1024 context tokens

  char* w = (char*)d_ws;
  size_t off = 0;
  auto alloc = [&](size_t bytes) -> char* {
    char* p = w + off;
    off = (off + bytes + 255) & ~(size_t)255;
    return p;
  };

  float* xcur = (float*)alloc((size_t)T * 512 * 4);       // running activation
  __bf16* hb = (__bf16*)alloc((size_t)T * 512 * 2);       // LN output
  __bf16* qb = (__bf16*)alloc((size_t)T * 512 * 2);
  __bf16* kb = (__bf16*)alloc((size_t)T * 512 * 2);
  __bf16* vt = (__bf16*)alloc((size_t)Bb * 512 * Nn * 2); // V transposed
  __bf16* ao = (__bf16*)alloc((size_t)T * 512 * 2);       // attn out
  __bf16* ctxt = (__bf16*)alloc((size_t)Tc * CTXC * 2);   // ctx token-major
  __bf16* act = (__bf16*)alloc((size_t)T * 2048 * 2);     // geglu act
  __bf16* wq1 = (__bf16*)alloc(512 * 512 * 2);
  __bf16* wk1 = (__bf16*)alloc(512 * 512 * 2);
  __bf16* wv1 = (__bf16*)alloc(512 * 512 * 2);
  __bf16* wo1 = (__bf16*)alloc(512 * 512 * 2);
  __bf16* wq2 = (__bf16*)alloc(512 * 512 * 2);
  __bf16* wk2 = (__bf16*)alloc(512 * 768 * 2);
  __bf16* wv2 = (__bf16*)alloc(512 * 768 * 2);
  __bf16* wo2 = (__bf16*)alloc(512 * 512 * 2);
  __bf16* wf1 = (__bf16*)alloc((size_t)4096 * 512 * 2);
  __bf16* wf2 = (__bf16*)alloc((size_t)512 * 2048 * 2);

  auto conv = [&](const float* s, __bf16* d, int n) {
    f32_to_bf16<<<dim3((n + 255) / 256), dim3(256), 0, stream>>>(s, d, n);
  };
  conv(Wq1, wq1, 512 * 512);
  conv(Wk1, wk1, 512 * 512);
  conv(Wv1, wv1, 512 * 512);
  conv(Wo1, wo1, 512 * 512);
  conv(Wq2, wq2, 512 * 512);
  conv(Wk2, wk2, 512 * 768);
  conv(Wv2, wv2, 512 * 768);
  conv(Wo2, wo2, 512 * 512);
  conv(Wff1, wf1, 4096 * 512);
  conv(Wff2, wf2, 512 * 2048);

  // x [b,512,2048] -> xcur [8192,512] fp32 ; context -> ctxt bf16 [1024,768]
  transpose_bcn_f32<<<dim3(Nn / 32, 512 / 32, Bb), dim3(32, 8), 0, stream>>>(
      x, xcur, 512, Nn);
  transpose_bcn_bf16<<<dim3(Mm / 32, CTXC / 32, Bb), dim3(32, 8), 0, stream>>>(
      ctx, ctxt, CTXC, Mm);

  dim3 gB(256);
  dim3 gProj(T / 64, 512 / 128);       // 8192-token x 512-out GEMMs
  dim3 gProjC(Tc / 64, 512 / 128);     // 1024-token (context) GEMMs
  dim3 gFlash(Nn / 64, 8, Bb);

  // ---- self attention ----
  ln512_tokmajor<<<dim3(T / 8), gB, 0, stream>>>(xcur, g1, hb);
  gemm_bf16_out<<<gProj, gB, 0, stream>>>(hb, wq1, qb, 512, 512);
  gemm_bf16_out<<<gProj, gB, 0, stream>>>(hb, wk1, kb, 512, 512);
  gemm_vt<<<gProj, gB, 0, stream>>>(hb, wv1, vt, 512, Nn);
  flash_attn<<<gFlash, dim3(128), 0, stream>>>(qb, kb, vt, ao, Nn, Nn);
  gemm_f32_bias_res<<<gProj, gB, 0, stream>>>(ao, wo1, bo1, xcur, 512, 512);

  // ---- cross attention ----
  ln512_tokmajor<<<dim3(T / 8), gB, 0, stream>>>(xcur, g2, hb);
  gemm_bf16_out<<<gProj, gB, 0, stream>>>(hb, wq2, qb, 512, 512);
  gemm_bf16_out<<<gProjC, gB, 0, stream>>>(ctxt, wk2, kb, 512, CTXC);
  gemm_vt<<<gProjC, gB, 0, stream>>>(ctxt, wv2, vt, CTXC, Mm);
  flash_attn<<<gFlash, dim3(128), 0, stream>>>(qb, kb, vt, ao, Nn, Mm);
  gemm_f32_bias_res<<<gProj, gB, 0, stream>>>(ao, wo2, bo2, xcur, 512, 512);

  // ---- GEGLU feed-forward (fused) ----
  ln512_tokmajor<<<dim3(T / 8), gB, 0, stream>>>(xcur, g3, hb);
  gemm_geglu<<<dim3(T / 64, 2048 / 128), gB, 0, stream>>>(hb, wf1, bff1, act,
                                                          512);
  gemm_final<<<dim3(T / 64, 512 / 128), gB, 0, stream>>>(act, wf2, bff2, xcur,
                                                         OUT, 2048, Nn);
}
